// Model_DIN_V2_Gru_att_Gru_18176301597126
// MI455X (gfx1250) — compile-verified, matrix operand order checked by reading, not run on hardware
//
#include <hip/hip_runtime.h>
#include <hip/hip_bf16.h>

typedef __bf16 bf16;
typedef __attribute__((ext_vector_type(16))) __bf16 v16bf;
typedef __attribute__((ext_vector_type(8)))  float  v8f;

#define BB   1024
#define TT   200
#define HH   128
#define DINP 576
#define BN_SCALE 0.9995003746879549f

// ---------------- fast transcendentals (hardware TRANS ops) ----------------
__device__ inline float fast_rcp(float x) {
#if __has_builtin(__builtin_amdgcn_rcpf)
  return __builtin_amdgcn_rcpf(x);
#else
  return 1.0f / x;
#endif
}
__device__ inline float fast_sigmoid(float x) {
  return fast_rcp(1.0f + __expf(-x));
}
__device__ inline float fast_tanh(float x) {
#if __has_builtin(__builtin_amdgcn_tanhf)
  return __builtin_amdgcn_tanhf(x);
#else
  return 2.0f * fast_sigmoid(2.0f * x) - 1.0f;
#endif
}

// ---------------- WMMA fragment helpers (CDNA5 16x16x32 bf16 layouts) ----------
// A (16x32, row-major src, stride ld): lane -> row m = lane&15, half = lane>>4.
// element j (j<8 -> k=j, j>=8 -> k=j+8), plus half*8.
__device__ inline v16bf load_frag_a(const bf16* __restrict__ base, int ld, int lane) {
  int m = lane & 15, half = lane >> 4;
  const bf16* p = base + m * ld + half * 8;
  v16bf a;
#pragma unroll
  for (int j = 0; j < 16; ++j) {
    int k = (j < 8) ? j : (j + 8);
    a[j] = p[k];
  }
  return a;
}
// B (32x16, row-major KxN src, stride ld): lane -> col n = lane&15, same k map.
__device__ inline v16bf load_frag_b(const bf16* __restrict__ base, int ld, int lane) {
  int n = lane & 15, half = lane >> 4;
  const bf16* p = base + n + (size_t)(half * 8) * ld;
  v16bf b;
#pragma unroll
  for (int j = 0; j < 16; ++j) {
    int k = (j < 8) ? j : (j + 8);
    b[j] = p[(size_t)k * ld];
  }
  return b;
}
__device__ inline v8f wmma_bf16(v16bf a, v16bf b, v8f c) {
  return __builtin_amdgcn_wmma_f32_16x16x32_bf16(false, a, false, b, (short)0, c, false, false);
}

// ---------------- weight fp32 -> bf16 ----------------
__global__ void f2bf_kernel(const float* __restrict__ src, bf16* __restrict__ dst, int n) {
  int i = blockIdx.x * blockDim.x + threadIdx.x;
  if (i < n) dst[i] = (bf16)src[i];
}

// ---------------- embedding gather + his_sum + bf16 history ----------------
__global__ __launch_bounds__(128)
void embed_kernel(const int* __restrict__ uid, const int* __restrict__ mid, const int* __restrict__ cat,
                  const int* __restrict__ mid_his, const int* __restrict__ cat_his,
                  const float* __restrict__ uid_emb, const float* __restrict__ mid_emb,
                  const float* __restrict__ cat_emb,
                  float* __restrict__ uid_e, float* __restrict__ item_eb,
                  float* __restrict__ his_sum, bf16* __restrict__ xh) {
  int b = blockIdx.x, c = threadIdx.x;
  if (c < 64) uid_e[b * 64 + c] = uid_emb[(size_t)uid[b] * 64 + c];
  float ie = (c < 64) ? mid_emb[(size_t)mid[b] * 64 + c] : cat_emb[(size_t)cat[b] * 64 + (c - 64)];
  item_eb[b * HH + c] = ie;
  float hs = 0.f;
  for (int t = 0; t < TT; ++t) {
    int im = mid_his[b * TT + t], ic = cat_his[b * TT + t];
    float x = (c < 64) ? mid_emb[(size_t)im * 64 + c] : cat_emb[(size_t)ic * 64 + (c - 64)];
    hs += x;
    xh[((size_t)b * TT + t) * HH + c] = (bf16)x;
  }
  his_sum[b * HH + c] = hs;
}

// ---------------- persistent GRU: one block = 16 batch rows, 8 waves ----------
// gates = sigmoid([x,h]@Wg + bg) -> r,u ; c = tanh([x, r*h]@Wc + bc) ; h = u*h+(1-u)*c
__global__ __launch_bounds__(256)
void gru_kernel(const bf16* __restrict__ xin,   // (B,T,128) bf16
                const bf16* __restrict__ Wg,    // (256,256) bf16
                const float* __restrict__ bg,   // (256)
                const bf16* __restrict__ Wc,    // (256,128) bf16
                const float* __restrict__ bc,   // (128)
                float* __restrict__ seq_out,    // (B,T,128) f32 or nullptr
                float* __restrict__ final_out)  // (B,128) or nullptr
{
  __shared__ bf16  sX [16 * HH];
  __shared__ bf16  sH [16 * HH];
  __shared__ float fH [16 * HH];
  __shared__ bf16  sRH[16 * HH];
  __shared__ float sU [16 * HH];

  int tid = threadIdx.x, lane = tid & 31, wave = tid >> 5;
  int bbase = blockIdx.x * 16;

  for (int i = tid; i < 16 * HH; i += 256) { fH[i] = 0.f; sH[i] = (bf16)0.f; }
  __syncthreads();

  for (int t = 0; t < TT; ++t) {
    for (int i = tid; i < 16 * HH; i += 256) {
      int r = i >> 7, c = i & 127;
      sX[i] = xin[((size_t)(bbase + r) * TT + t) * HH + c];
    }
    // prefetch next timestep's x tile (global_prefetch_b8): 16 rows x 2 cachelines
    if (t + 1 < TT && tid < 32) {
      __builtin_prefetch(xin + ((size_t)(bbase + (tid >> 1)) * TT + (t + 1)) * HH + (tid & 1) * 64, 0, 0);
    }
    __syncthreads();

    // gates: 16x256 output, 16 N-tiles, 2 per wave
#pragma unroll
    for (int nti = 0; nti < 2; ++nti) {
      int nt = wave * 2 + nti;
      v8f acc = {};
#pragma unroll
      for (int kk = 0; kk < 8; ++kk) {
        const bf16* ab = (kk < 4) ? (sX + kk * 32) : (sH + (kk - 4) * 32);
        v16bf af = load_frag_a(ab, HH, lane);
        v16bf bf_ = load_frag_b(Wg + (size_t)kk * 32 * 256 + nt * 16, 256, lane);
        acc = wmma_bf16(af, bf_, acc);
      }
      int n = lane & 15, half = lane >> 4;
      int nc = nt * 16 + n;
      if (nt < 8) {
        // r-gate tiles (wave-uniform branch, full-EXEC stores)
#pragma unroll
        for (int r = 0; r < 8; ++r) {
          int m = r + 8 * half;
          float g = fast_sigmoid(acc[r] + bg[nc]);
          sRH[m * HH + nc] = (bf16)(g * fH[m * HH + nc]);            // r*h
        }
      } else {
        // u-gate tiles
#pragma unroll
        for (int r = 0; r < 8; ++r) {
          int m = r + 8 * half;
          float g = fast_sigmoid(acc[r] + bg[nc]);
          sU[m * HH + (nc - HH)] = g;                                // u
        }
      }
    }
    __syncthreads();

    // candidate + state update: 16x128, 1 N-tile per wave
    {
      int nt = wave;
      v8f acc = {};
#pragma unroll
      for (int kk = 0; kk < 8; ++kk) {
        const bf16* ab = (kk < 4) ? (sX + kk * 32) : (sRH + (kk - 4) * 32);
        v16bf af = load_frag_a(ab, HH, lane);
        v16bf bf_ = load_frag_b(Wc + (size_t)kk * 32 * HH + nt * 16, HH, lane);
        acc = wmma_bf16(af, bf_, acc);
      }
      int n = lane & 15, half = lane >> 4;
      int nc = nt * 16 + n;
#pragma unroll
      for (int r = 0; r < 8; ++r) {
        int m = r + 8 * half;
        float c = fast_tanh(acc[r] + bc[nc]);
        float u = sU[m * HH + nc];
        float hn = u * fH[m * HH + nc] + (1.f - u) * c;
        fH[m * HH + nc] = hn;
        sH[m * HH + nc] = (bf16)hn;
        if (seq_out) seq_out[((size_t)(bbase + m) * TT + t) * HH + nc] = hn;
      }
    }
    __syncthreads();
  }

  if (final_out) {
    for (int i = tid; i < 16 * HH; i += 256) {
      int r = i >> 7, c = i & 127;
      final_out[(size_t)(bbase + r) * HH + c] = fH[i];
    }
  }
}

// ---------------- q = prelu(item_eb @ Wq + bq, aq) ----------------
__global__ __launch_bounds__(128)
void q_kernel(const float* __restrict__ item_eb, const float* __restrict__ Wq,
              const float* __restrict__ bq, const float* __restrict__ aq,
              float* __restrict__ q) {
  int b = blockIdx.x, n = threadIdx.x;
  float s = bq[n];
  for (int k = 0; k < HH; ++k) s += item_eb[b * HH + k] * Wq[k * HH + n];
  q[b * HH + n] = fmaxf(s, 0.f) + aq[n] * fminf(s, 0.f);
}

// ---------------- fused attention MLP: din_all -> 80 (WMMA) -> 40 -> score ----
__global__ __launch_bounds__(256)
void att_fcn_kernel(const float* __restrict__ q, const float* __restrict__ rnn1,
                    const bf16* __restrict__ Wa1, const float* __restrict__ ba1,
                    const float* __restrict__ Wa2, const float* __restrict__ ba2,
                    const float* __restrict__ Wa3, const float* __restrict__ ba3,
                    const float* __restrict__ mask, float* __restrict__ scores) {
  __shared__ bf16  sDin[16 * 512];
  __shared__ float sA1 [16 * 80];
  __shared__ float sA2 [16 * 40];
  int tid = threadIdx.x, lane = tid & 31, wave = tid >> 5;
  long rowbase = (long)blockIdx.x * 16;

  for (int i = tid; i < 16 * 512; i += 256) {
    int r = i >> 9, c = i & 511;
    long row = rowbase + r;
    int b = (int)(row / TT);
    int cc = c & 127, reg = c >> 7;
    float qv = q[b * HH + cc];
    float rv = rnn1[(size_t)row * HH + cc];
    float v = (reg == 0) ? qv : (reg == 1) ? rv : (reg == 2) ? (qv - rv) : (qv * rv);
    sDin[i] = (bf16)v;
  }
  __syncthreads();

  if (wave < 5) {  // 80 = 5 N-tiles of 16
    v8f acc = {};
#pragma unroll
    for (int kk = 0; kk < 16; ++kk) {
      v16bf af = load_frag_a(sDin + kk * 32, 512, lane);
      v16bf bf_ = load_frag_b(Wa1 + (size_t)kk * 32 * 80 + wave * 16, 80, lane);
      acc = wmma_bf16(af, bf_, acc);
    }
    int n = lane & 15, half = lane >> 4;
    int nc = wave * 16 + n;
#pragma unroll
    for (int r = 0; r < 8; ++r) {
      int m = r + 8 * half;
      sA1[m * 80 + nc] = fast_sigmoid(acc[r] + ba1[nc]);
    }
  }
  __syncthreads();

  for (int i = tid; i < 16 * 40; i += 256) {
    int r = i / 40, n = i % 40;
    float s = ba2[n];
    for (int k = 0; k < 80; ++k) s += sA1[r * 80 + k] * Wa2[k * 40 + n];
    sA2[i] = fast_sigmoid(s);
  }
  __syncthreads();

  if (tid < 16) {
    long row = rowbase + tid;
    float s = ba3[0];
    for (int k = 0; k < 40; ++k) s += sA2[tid * 40 + k] * Wa3[k];
    scores[row] = (mask[row] == 1.0f) ? s : (-4294967296.0f + 1.0f);
  }
}

// ---------------- softmax over T + att_out = rnn1 * alpha (bf16) -------------
__global__ __launch_bounds__(256)
void softmax_attout_kernel(const float* __restrict__ scores, const float* __restrict__ rnn1,
                           bf16* __restrict__ attout) {
  __shared__ float red[256];
  __shared__ float sAl[TT];
  int b = blockIdx.x, tid = threadIdx.x;
  float v = (tid < TT) ? scores[b * TT + tid] : -3.4e38f;
  red[tid] = v; __syncthreads();
  for (int s = 128; s > 0; s >>= 1) { if (tid < s) red[tid] = fmaxf(red[tid], red[tid + s]); __syncthreads(); }
  float mx = red[0]; __syncthreads();
  float e = (tid < TT) ? __expf(v - mx) : 0.f;
  red[tid] = e; __syncthreads();
  for (int s = 128; s > 0; s >>= 1) { if (tid < s) red[tid] += red[tid + s]; __syncthreads(); }
  float inv = fast_rcp(red[0]);
  if (tid < TT) sAl[tid] = e * inv;
  __syncthreads();
  for (int i = tid; i < TT * HH; i += 256) {
    int t = i >> 7, c = i & 127;
    size_t off = ((size_t)b * TT + t) * HH + c;
    attout[off] = (bf16)(rnn1[off] * sAl[t]);
  }
}

// ---------------- concat + inference-BN ----------------
__global__ void bn_kernel(const float* __restrict__ uid_e, const float* __restrict__ item_eb,
                          const float* __restrict__ his_sum, const float* __restrict__ final2,
                          const float* __restrict__ bn_g, const float* __restrict__ bn_b,
                          float* __restrict__ bn) {
  int idx = blockIdx.x * blockDim.x + threadIdx.x;
  if (idx >= BB * DINP) return;
  int b = idx / DINP, j = idx % DINP;
  float v;
  if      (j < 64)  v = uid_e[b * 64 + j];
  else if (j < 192) v = item_eb[b * HH + (j - 64)];
  else if (j < 320) v = his_sum[b * HH + (j - 192)];
  else if (j < 448) v = item_eb[b * HH + (j - 320)] * his_sum[b * HH + (j - 320)];
  else              v = final2[b * HH + (j - 448)];
  bn[idx] = bn_g[j] * v * BN_SCALE + bn_b[j];
}

// ---------------- generic FC (small) ----------------
__global__ void fc_kernel(const float* __restrict__ X, const float* __restrict__ W,
                          const float* __restrict__ bias, float* __restrict__ Z,
                          int Bn, int K, int N) {
  int idx = blockIdx.x * blockDim.x + threadIdx.x;
  if (idx >= Bn * N) return;
  int b = idx / N, n = idx % N;
  float s = bias[n];
  for (int k = 0; k < K; ++k) s += X[(size_t)b * K + k] * W[(size_t)k * N + n];
  Z[idx] = s;
}

// ---------------- dice batch-statistics + apply ----------------
__global__ __launch_bounds__(256)
void colstats_kernel(const float* __restrict__ Z, float* __restrict__ mean,
                     float* __restrict__ stdv, int Bn, int N) {
  __shared__ float s1[256], s2[256];
  int d = blockIdx.x, tid = threadIdx.x;
  float a = 0.f, b2 = 0.f;
  for (int r = tid; r < Bn; r += 256) { float v = Z[(size_t)r * N + d]; a += v; b2 += v * v; }
  s1[tid] = a; s2[tid] = b2; __syncthreads();
  for (int s = 128; s > 0; s >>= 1) {
    if (tid < s) { s1[tid] += s1[tid + s]; s2[tid] += s2[tid + s]; }
    __syncthreads();
  }
  if (tid == 0) {
    float m = s1[0] / Bn;
    float var = s2[0] / Bn - m * m;
    mean[d] = m;
    stdv[d] = sqrtf(var + 1e-9f);
  }
}

__global__ void dice_kernel(const float* __restrict__ Z, const float* __restrict__ mean,
                            const float* __restrict__ stdv, const float* __restrict__ alpha,
                            float* __restrict__ Y, int Bn, int N) {
  int idx = blockIdx.x * blockDim.x + threadIdx.x;
  if (idx >= Bn * N) return;
  int n = idx % N;
  float x = Z[idx];
  float xp = fast_sigmoid((x - mean[n]) * fast_rcp(stdv[n] + 1e-9f));
  Y[idx] = alpha[n] * (1.f - xp) * x + xp * x;
}

// ---------------- final 80->2 + softmax + 1e-8 ----------------
__global__ void out_kernel(const float* __restrict__ d2, const float* __restrict__ Wf3,
                           const float* __restrict__ bf3, float* __restrict__ out) {
  int b = blockIdx.x * blockDim.x + threadIdx.x;
  if (b >= BB) return;
  float l0 = bf3[0], l1 = bf3[1];
  for (int k = 0; k < 80; ++k) { float v = d2[b * 80 + k]; l0 += v * Wf3[k * 2]; l1 += v * Wf3[k * 2 + 1]; }
  float m = fmaxf(l0, l1);
  float e0 = __expf(l0 - m), e1 = __expf(l1 - m);
  float inv = fast_rcp(e0 + e1);
  out[b * 2 + 0] = e0 * inv + 1e-8f;
  out[b * 2 + 1] = e1 * inv + 1e-8f;
}

extern "C" void kernel_launch(void* const* d_in, const int* in_sizes, int n_in,
                              void* d_out, int out_size, void* d_ws, size_t ws_size,
                              hipStream_t stream) {
  (void)in_sizes; (void)n_in; (void)out_size; (void)ws_size;
  const int*   uid     = (const int*)d_in[0];
  const int*   mid     = (const int*)d_in[1];
  const int*   cat     = (const int*)d_in[2];
  const int*   mid_his = (const int*)d_in[3];
  const int*   cat_his = (const int*)d_in[4];
  const float* mask    = (const float*)d_in[5];
  const float* uid_emb = (const float*)d_in[6];
  const float* mid_emb = (const float*)d_in[7];
  const float* cat_emb = (const float*)d_in[8];
  const float* Wg1 = (const float*)d_in[9];  const float* bg1 = (const float*)d_in[10];
  const float* Wc1 = (const float*)d_in[11]; const float* bc1 = (const float*)d_in[12];
  const float* Wq  = (const float*)d_in[13]; const float* bq  = (const float*)d_in[14];
  const float* aq  = (const float*)d_in[15];
  const float* Wa1 = (const float*)d_in[16]; const float* ba1 = (const float*)d_in[17];
  const float* Wa2 = (const float*)d_in[18]; const float* ba2 = (const float*)d_in[19];
  const float* Wa3 = (const float*)d_in[20]; const float* ba3 = (const float*)d_in[21];
  const float* Wg2 = (const float*)d_in[22]; const float* bg2 = (const float*)d_in[23];
  const float* Wc2 = (const float*)d_in[24]; const float* bc2 = (const float*)d_in[25];
  const float* bn_g = (const float*)d_in[26]; const float* bn_b = (const float*)d_in[27];
  const float* Wf1 = (const float*)d_in[28]; const float* bf1 = (const float*)d_in[29];
  const float* ad1 = (const float*)d_in[30];
  const float* Wf2 = (const float*)d_in[31]; const float* bf2 = (const float*)d_in[32];
  const float* ad2 = (const float*)d_in[33];
  const float* Wf3 = (const float*)d_in[34]; const float* bf3 = (const float*)d_in[35];
  float* out = (float*)d_out;

  char* ws = (char*)d_ws;
  size_t off = 0;
  auto alloc = [&](size_t bytes) -> char* {
    char* p = ws + off;
    off += (bytes + 255) & ~(size_t)255;
    return p;
  };

  bf16*  xh      = (bf16*) alloc((size_t)BB * TT * HH * 2);   // history bf16
  float* rnn1    = (float*)alloc((size_t)BB * TT * HH * 4);   // GRU1 outputs
  bf16*  attout  = (bf16*) alloc((size_t)BB * TT * HH * 2);   // GRU2 inputs
  float* uid_e   = (float*)alloc((size_t)BB * 64 * 4);
  float* item_eb = (float*)alloc((size_t)BB * HH * 4);
  float* his_sum = (float*)alloc((size_t)BB * HH * 4);
  float* qbuf    = (float*)alloc((size_t)BB * HH * 4);
  float* scores  = (float*)alloc((size_t)BB * TT * 4);
  float* final2  = (float*)alloc((size_t)BB * HH * 4);
  float* bnbuf   = (float*)alloc((size_t)BB * DINP * 4);
  float* z1      = (float*)alloc((size_t)BB * 200 * 4);
  float* d1      = (float*)alloc((size_t)BB * 200 * 4);
  float* z2      = (float*)alloc((size_t)BB * 80 * 4);
  float* d2      = (float*)alloc((size_t)BB * 80 * 4);
  float* mean1   = (float*)alloc(200 * 4);
  float* std1    = (float*)alloc(200 * 4);
  float* mean2   = (float*)alloc(80 * 4);
  float* std2    = (float*)alloc(80 * 4);
  bf16*  Wg1b    = (bf16*) alloc(256 * 256 * 2);
  bf16*  Wc1b    = (bf16*) alloc(256 * 128 * 2);
  bf16*  Wg2b    = (bf16*) alloc(256 * 256 * 2);
  bf16*  Wc2b    = (bf16*) alloc(256 * 128 * 2);
  bf16*  Wa1b    = (bf16*) alloc(512 * 80 * 2);

  // weights -> bf16
  f2bf_kernel<<<256, 256, 0, stream>>>(Wg1, Wg1b, 256 * 256);
  f2bf_kernel<<<128, 256, 0, stream>>>(Wc1, Wc1b, 256 * 128);
  f2bf_kernel<<<256, 256, 0, stream>>>(Wg2, Wg2b, 256 * 256);
  f2bf_kernel<<<128, 256, 0, stream>>>(Wc2, Wc2b, 256 * 128);
  f2bf_kernel<<<160, 256, 0, stream>>>(Wa1, Wa1b, 512 * 80);

  // gathers
  embed_kernel<<<BB, 128, 0, stream>>>(uid, mid, cat, mid_his, cat_his,
                                       uid_emb, mid_emb, cat_emb,
                                       uid_e, item_eb, his_sum, xh);

  // GRU1 (persistent, WMMA): writes rnn1
  gru_kernel<<<BB / 16, 256, 0, stream>>>(xh, Wg1b, bg1, Wc1b, bc1, rnn1, nullptr);

  // attention
  q_kernel<<<BB, 128, 0, stream>>>(item_eb, Wq, bq, aq, qbuf);
  att_fcn_kernel<<<(BB * TT) / 16, 256, 0, stream>>>(qbuf, rnn1, Wa1b, ba1,
                                                     Wa2, ba2, Wa3, ba3, mask, scores);
  softmax_attout_kernel<<<BB, 256, 0, stream>>>(scores, rnn1, attout);

  // GRU2 (persistent, WMMA): final state only
  gru_kernel<<<BB / 16, 256, 0, stream>>>(attout, Wg2b, bg2, Wc2b, bc2, nullptr, final2);

  // head
  bn_kernel<<<(BB * DINP + 255) / 256, 256, 0, stream>>>(uid_e, item_eb, his_sum, final2,
                                                         bn_g, bn_b, bnbuf);
  fc_kernel<<<(BB * 200 + 255) / 256, 256, 0, stream>>>(bnbuf, Wf1, bf1, z1, BB, DINP, 200);
  colstats_kernel<<<200, 256, 0, stream>>>(z1, mean1, std1, BB, 200);
  dice_kernel<<<(BB * 200 + 255) / 256, 256, 0, stream>>>(z1, mean1, std1, ad1, d1, BB, 200);
  fc_kernel<<<(BB * 80 + 255) / 256, 256, 0, stream>>>(d1, Wf2, bf2, z2, BB, 200, 80);
  colstats_kernel<<<80, 256, 0, stream>>>(z2, mean2, std2, BB, 80);
  dice_kernel<<<(BB * 80 + 255) / 256, 256, 0, stream>>>(z2, mean2, std2, ad2, d2, BB, 80);
  out_kernel<<<(BB + 255) / 256, 256, 0, stream>>>(d2, Wf3, bf3, out);
}
